// BitLinear_34291018892051
// MI455X (gfx1250) — matrix operand order repaired and checked by reading, not compiled, not run on hardware
//
#include <hip/hip_runtime.h>
#include <hip/hip_bf16.h>
#include <math.h>

typedef __attribute__((ext_vector_type(16))) _Float16 v16h;
typedef __attribute__((ext_vector_type(8)))  _Float16 v8h;
typedef __attribute__((ext_vector_type(4)))  _Float16 v4h;
typedef __attribute__((ext_vector_type(8)))  float    v8f;
typedef __attribute__((ext_vector_type(4)))  float    v4f;

static constexpr int   MDIM      = 8192;   // B * S
static constexpr int   KDIM      = 2048;   // D_IN
static constexpr int   NDIM      = 2048;   // D_OUT
static constexpr float NORM_EPS  = 1e-6f;
static constexpr float QUANT_EPS = 1e-8f;

union frag16 { v16h v; v8h h[2]; };

// ---------- kernel 1: per-block partial sums of |w| (deterministic) ----------
__global__ void k_abs_partial(const float* __restrict__ w,
                              float* __restrict__ partial, int n4) {
  __shared__ float sm[256];
  const int tid = threadIdx.x;
  float s = 0.f;
  for (int i = blockIdx.x * 256 + tid; i < n4; i += gridDim.x * 256) {
    v4f v = ((const v4f*)w)[i];
    s += __builtin_fabsf(v.x) + __builtin_fabsf(v.y) +
         __builtin_fabsf(v.z) + __builtin_fabsf(v.w);
  }
  sm[tid] = s;
  __syncthreads();
  for (int off = 128; off > 0; off >>= 1) {
    if (tid < off) sm[tid] += sm[tid + off];
    __syncthreads();
  }
  if (tid == 0) partial[blockIdx.x] = sm[0];
}

// ---------- kernel 2: finalize gamma = mean(|w|) ----------
__global__ void k_gamma(const float* __restrict__ partial,
                        float* __restrict__ gamma, int nparts, float inv_count) {
  __shared__ float sm[256];
  const int tid = threadIdx.x;
  float s = 0.f;
  for (int i = tid; i < nparts; i += 256) s += partial[i];
  sm[tid] = s;
  __syncthreads();
  for (int off = 128; off > 0; off >>= 1) {
    if (tid < off) sm[tid] += sm[tid + off];
    __syncthreads();
  }
  if (tid == 0) gamma[0] = sm[0] * inv_count;
}

// ---------- kernel 3: ternary quantize weights -> f16, same [N,K] layout ----------
__global__ void k_quant(const float* __restrict__ w, _Float16* __restrict__ wq,
                        const float* __restrict__ gamma_p, int n4) {
  const int i = blockIdx.x * 256 + threadIdx.x;
  if (i >= n4) return;
  const float inv = 1.f / (gamma_p[0] + QUANT_EPS);
  v4f v = ((const v4f*)w)[i];
  v4h o;
  o[0] = (_Float16)fminf(fmaxf(__builtin_rintf(v.x * inv), -1.f), 1.f);
  o[1] = (_Float16)fminf(fmaxf(__builtin_rintf(v.y * inv), -1.f), 1.f);
  o[2] = (_Float16)fminf(fmaxf(__builtin_rintf(v.z * inv), -1.f), 1.f);
  o[3] = (_Float16)fminf(fmaxf(__builtin_rintf(v.w * inv), -1.f), 1.f);
  ((v4h*)wq)[i] = o;
}

// ---------- kernel 4: RMSNorm one row per block -> f16 activations ----------
__global__ void k_rmsnorm(const float* __restrict__ x, const float* __restrict__ g,
                          _Float16* __restrict__ xn) {
  __shared__ float sm[256];
  const int row = blockIdx.x, tid = threadIdx.x;
  const float* xr = x + (size_t)row * KDIM;
  v4f a = ((const v4f*)xr)[tid * 2];
  v4f b = ((const v4f*)xr)[tid * 2 + 1];
  float ss = a.x*a.x + a.y*a.y + a.z*a.z + a.w*a.w +
             b.x*b.x + b.y*b.y + b.z*b.z + b.w*b.w;
  sm[tid] = ss;
  __syncthreads();
  for (int off = 128; off > 0; off >>= 1) {
    if (tid < off) sm[tid] += sm[tid + off];
    __syncthreads();
  }
  const float mean = sm[0] * (1.0f / (float)KDIM);
  const float rinv = 1.0f / sqrtf(mean + NORM_EPS);
  const float* gp = g + tid * 8;
  v8h o;
  o[0] = (_Float16)(a.x * rinv * gp[0]);
  o[1] = (_Float16)(a.y * rinv * gp[1]);
  o[2] = (_Float16)(a.z * rinv * gp[2]);
  o[3] = (_Float16)(a.w * rinv * gp[3]);
  o[4] = (_Float16)(b.x * rinv * gp[4]);
  o[5] = (_Float16)(b.y * rinv * gp[5]);
  o[6] = (_Float16)(b.z * rinv * gp[6]);
  o[7] = (_Float16)(b.w * rinv * gp[7]);
  ((v8h*)(xn + (size_t)row * KDIM))[tid] = o;
}

// ---------- kernel 5: WMMA GEMM  out[m,n] = sum_k A[m,k]*Bq[n,k] * gamma ----------
// 8 waves/block arranged 2(M) x 4(N); each wave: 32x64 tile = 2x4 WMMA frags
// (8 v_wmma per K-step of 32, 12 fragment loads -> 0.67 wmma/load).
__global__ __launch_bounds__(256)
void k_gemm(const _Float16* __restrict__ A, const _Float16* __restrict__ Bq,
            const float* __restrict__ gamma_p, float* __restrict__ out) {
  const int lane = threadIdx.x & 31;
  const int wave = threadIdx.x >> 5;
  const int wm = wave & 1;          // 0..1
  const int wn = wave >> 1;         // 0..3
  const int m0 = blockIdx.y * 64 + wm * 32;
  const int n0 = blockIdx.x * 256 + wn * 64;
  const int l15 = lane & 15;
  const int hi = lane >> 4;         // 0: lanes 0-15, 1: lanes 16-31

  v8f acc[2][4] = {};

  const _Float16* arow0 = A + (size_t)(m0 + l15) * KDIM;
  const _Float16* arow1 = A + (size_t)(m0 + 16 + l15) * KDIM;
  const _Float16* bcol[4];
  #pragma unroll
  for (int j = 0; j < 4; ++j)
    bcol[j] = Bq + (size_t)(n0 + j * 16 + l15) * KDIM;

  // A 16x32 f16 layout: lanes 0-15 hold K {0..7, 16..23}; lanes 16-31 {8..15, 24..31}
  const int aoff = hi * 8;
  // B 32x16 f16 layout: lane = n; lanes 0-15 hold K {0..7, 16..23}; lanes 16-31 {8..15, 24..31}
  const int boff = hi * 8;

  #pragma unroll 2
  for (int kb = 0; kb < KDIM; kb += 32) {
    frag16 a0, a1, b[4];
    a0.h[0] = *(const v8h*)(arow0 + kb + aoff);
    a0.h[1] = *(const v8h*)(arow0 + kb + aoff + 16);
    a1.h[0] = *(const v8h*)(arow1 + kb + aoff);
    a1.h[1] = *(const v8h*)(arow1 + kb + aoff + 16);
    #pragma unroll
    for (int j = 0; j < 4; ++j) {
      b[j].h[0] = *(const v8h*)(bcol[j] + kb + boff);
      b[j].h[1] = *(const v8h*)(bcol[j] + kb + boff + 16);
    }
    #pragma unroll
    for (int j = 0; j < 4; ++j) {
      acc[0][j] = __builtin_amdgcn_wmma_f32_16x16x32_f16(false, a0.v, false, b[j].v,
                                                         (short)0, acc[0][j], false, false);
      acc[1][j] = __builtin_amdgcn_wmma_f32_16x16x32_f16(false, a1.v, false, b[j].v,
                                                         (short)0, acc[1][j], false, false);
    }
  }

  const float gm = gamma_p[0];
  // C/D layout: VGPR r -> M = r (lanes 0-15) or M = 8+r (lanes 16-31); N = lane&15
  const int rbase = hi * 8;
  #pragma unroll
  for (int i = 0; i < 2; ++i) {
    #pragma unroll
    for (int j = 0; j < 4; ++j) {
      float* orow = out + (size_t)(m0 + i * 16 + rbase) * NDIM + (n0 + j * 16 + l15);
      #pragma unroll
      for (int r = 0; r < 8; ++r)
        orow[(size_t)r * NDIM] = acc[i][j][r] * gm;
    }
  }
}

extern "C" void kernel_launch(void* const* d_in, const int* in_sizes, int n_in,
                              void* d_out, int out_size, void* d_ws, size_t ws_size,
                              hipStream_t stream) {
  const float* x = (const float*)d_in[0];       // [2,4096,2048] fp32
  const float* w = (const float*)d_in[1];       // [2048,2048]  fp32
  const float* g = (const float*)d_in[2];       // [2048]       fp32
  float* out = (float*)d_out;                   // [2,4096,2048] fp32

  // workspace layout
  char* ws = (char*)d_ws;
  float*    gamma    = (float*)ws;                                       // 4 B
  float*    partials = (float*)(ws + 256);                               // 1024*4 B
  _Float16* xn16     = (_Float16*)(ws + 8192);                           // 32 MB
  _Float16* wq16     = (_Float16*)(ws + 8192 + (size_t)MDIM * KDIM * 2); // 8 MB

  const int n4 = (NDIM * KDIM) / 4;  // 1048576 vec4 weight elements

  k_abs_partial<<<1024, 256, 0, stream>>>(w, partials, n4);
  k_gamma<<<1, 256, 0, stream>>>(partials, gamma, 1024,
                                 1.0f / ((float)NDIM * (float)KDIM));
  k_quant<<<n4 / 256, 256, 0, stream>>>(w, wq16, gamma, n4);
  k_rmsnorm<<<MDIM, 256, 0, stream>>>(x, g, xn16);
  k_gemm<<<dim3(NDIM / 256, MDIM / 64), 256, 0, stream>>>(xn16, wq16, gamma, out);
}
